// Seq2SeqVAE_49495203119743
// MI455X (gfx1250) — matrix operand order repaired and checked by reading, not compiled, not run on hardware
//
#include <hip/hip_runtime.h>

// ---------------------------------------------------------------------------
// Seq2Seq VAE (encoder LSTM -> latent -> autoregressive decoder LSTM)
// Persistent batch-tiled kernel for gfx1250 (CDNA5, wave32, WMMA bf16).
// ---------------------------------------------------------------------------

typedef __bf16 bf16_t;
typedef __bf16 v16bf __attribute__((ext_vector_type(16)));
typedef float  v8f   __attribute__((ext_vector_type(8)));

#define B_TOT   1024
#define T_SRC   1024
#define TGT_LEN 1024
#define I_DIM   64
#define H_DIM   512
#define L_DIM   64
#define NGATE   2048         // 4*H
#define MB      32           // batch rows per workgroup
#define KENC    576          // H + I (concatenated weight K)
#define LDA     584          // padded LDS row stride (bf16 elems) for A/h buffer
#define CST     516          // padded LDS row stride (f32) for c state
#define ABUF    (MB * LDA)   // one A buffer, elems

// LDS layout (bytes)
#define OFF_A     0
#define OFF_C     (OFF_A    + 2 * ABUF * 2)      // double-buffered h||x (bf16)
#define OFF_BIAS  (OFF_C    + MB * CST * 4)      // 2048 f32 gate bias
#define OFF_WIH   (OFF_BIAS + NGATE * 4)         // decoder W_ih (2048 f32)
#define OFF_WOUT  (OFF_WIH  + NGATE * 4)         // fc_out weights (512 f32)
#define OFF_Z     (OFF_WOUT + H_DIM * 4)         // latent z (32x64 f32)
#define OFF_X     (OFF_Z    + MB * L_DIM * 4)    // decoder feedback x (32 f32)
#define OFF_RED   (OFF_X    + MB * 4)            // fc_out reduction (32 f32)
#define SMEM_BYTES (OFF_RED + MB * 4)            // = 167,680 bytes (< 320KB WGP LDS)

// ---------------------------------------------------------------------------

__device__ __forceinline__ float sigm(float x)   { return 1.0f / (1.0f + __expf(-x)); }
__device__ __forceinline__ float tanh_f(float x) { return 1.0f - 2.0f / (__expf(2.0f * x) + 1.0f); }

__device__ __forceinline__ v8f wmma_bf16(v16bf a, v16bf b, v8f c) {
    return __builtin_amdgcn_wmma_f32_16x16x32_bf16(false, a, false, b, (short)0, c, false, false);
}

union Frag32 { v16bf v; uint4 u[2]; };

// A (16x32 bf16) from LDS. ISA layout: lane<16: M=lane, K={kb..kb+7, kb+16..kb+23};
// lane>=16: M=lane-16, K={kb+8..kb+15, kb+24..kb+31}. Two 16B ds loads per lane.
__device__ __forceinline__ v16bf load_a_frag(const bf16_t* aBase, int mBase, int lane, int kb) {
    const int hi = lane >> 4;
    const int m  = mBase + (lane & 15);
    const bf16_t* row = aBase + m * LDA;
    Frag32 f;
    f.u[0] = *(const uint4*)(row + kb + hi * 8);
    f.u[1] = *(const uint4*)(row + kb + hi * 8 + 16);
    return f.v;
}

// B (32x16 bf16) from global, weights stored row-major W[n][k] (n = gate column).
// ISA layout: lane<16: N=lane, K={kb..kb+7, kb+16..kb+23}; lane>=16: N=lane-16, K={kb+8..kb+15, kb+24..kb+31}.
// Two 16B loads per lane.
__device__ __forceinline__ v16bf load_b_frag(const bf16_t* w, int nBase, int lane, int kb, int rs) {
    const int n = nBase + (lane & 15);
    const bf16_t* p = w + (size_t)n * rs + kb + ((lane >> 4) << 3);
    Frag32 f;
    f.u[0] = *(const uint4*)(p);
    f.u[1] = *(const uint4*)(p + 16);
    return f.v;
}

// Stage x_{t} (32 rows x 64 cols f32 -> bf16) into columns [512..575] of an A buffer.
__device__ __forceinline__ void load_x(const float* __restrict__ src, bf16_t* __restrict__ dst,
                                       int b0, int t) {
    for (int e = threadIdx.x; e < MB * 16; e += 256) {
        const int m  = e >> 4;
        const int c4 = e & 15;
        const float4 v = *(const float4*)(src + ((size_t)(b0 + m) * T_SRC + t) * I_DIM + c4 * 4);
        bf16_t* d = dst + m * LDA + H_DIM + c4 * 4;
        d[0] = (bf16_t)v.x; d[1] = (bf16_t)v.y; d[2] = (bf16_t)v.z; d[3] = (bf16_t)v.w;
    }
}

// One LSTM step for a 32-row batch tile. Each wave owns 4 hidden 16-col tiles;
// for each it accumulates i/f/g/o gate tiles for both 16-row M subtiles
// (B fragments reused across M), then applies the cell nonlinearity using the
// ISA C/D lane mapping (VGPR r, lane l -> M = r + 8*(l>=16), N = l&15).
template<int KTILES, bool DEC>
__device__ __forceinline__ void lstm_step(
    const bf16_t* __restrict__ W, int rs,
    const bf16_t* __restrict__ aCur, bf16_t* __restrict__ aNxt,
    float* __restrict__ cS, const float* __restrict__ biasS,
    const float* __restrict__ wihS, const float* __restrict__ xS,
    int wave, int lane)
{
    const int hi = lane >> 4;
    const int nn = lane & 15;
    for (int q = 0; q < 4; ++q) {
        const int ht = (wave << 2) + q;          // hidden 16-col tile index (0..31)
        v8f acc[8];                               // [mt*4 + gate]
        #pragma unroll
        for (int i = 0; i < 8; ++i) {
            #pragma unroll
            for (int e = 0; e < 8; ++e) acc[i][e] = 0.0f;
        }

        for (int kk = 0; kk < KTILES; ++kk) {
            const int kb = kk << 5;
            const v16bf a0 = load_a_frag(aCur, 0,  lane, kb);
            const v16bf a1 = load_a_frag(aCur, 16, lane, kb);
            #pragma unroll
            for (int g = 0; g < 4; ++g) {
                const v16bf bfr = load_b_frag(W, (g << 9) + (ht << 4), lane, kb, rs);
                acc[g]     = wmma_bf16(a0, bfr, acc[g]);
                acc[4 + g] = wmma_bf16(a1, bfr, acc[4 + g]);
            }
        }

        const int j = (ht << 4) + nn;            // hidden index 0..511
        const float bi = biasS[j],        bff = biasS[512 + j];
        const float bg = biasS[1024 + j], bo  = biasS[1536 + j];
        float wxi = 0.f, wxf = 0.f, wxg = 0.f, wxo = 0.f;
        if constexpr (DEC) {
            wxi = wihS[j]; wxf = wihS[512 + j]; wxg = wihS[1024 + j]; wxo = wihS[1536 + j];
        }
        #pragma unroll
        for (int mt = 0; mt < 2; ++mt) {
            #pragma unroll
            for (int r = 0; r < 8; ++r) {
                const int m = (mt << 4) + (hi << 3) + r;   // batch row 0..31
                float gi = acc[mt * 4 + 0][r] + bi;
                float gf = acc[mt * 4 + 1][r] + bff;
                float gg = acc[mt * 4 + 2][r] + bg;
                float go = acc[mt * 4 + 3][r] + bo;
                if constexpr (DEC) {
                    const float xv = xS[m];                 // rank-1 input term (O==1)
                    gi += xv * wxi; gf += xv * wxf; gg += xv * wxg; go += xv * wxo;
                }
                const float cOld = cS[m * CST + j];
                const float cNew = sigm(gf) * cOld + sigm(gi) * tanh_f(gg);
                const float hNew = sigm(go) * tanh_f(cNew);
                cS[m * CST + j]   = cNew;
                aNxt[m * LDA + j] = (bf16_t)hNew;
            }
        }
    }
}

// ---------------------------------------------------------------------------
// Prep: pack encoder [W_hh | W_ih] into one bf16 (2048 x 576) matrix and
// convert decoder W_hh to bf16 (2048 x 512), both into workspace.
// ---------------------------------------------------------------------------
extern "C" __global__ void __launch_bounds__(256)
vae_prep(const float* __restrict__ Wih_e, const float* __restrict__ Whh_e,
         const float* __restrict__ Whh_d,
         bf16_t* __restrict__ wcat, bf16_t* __restrict__ wdec)
{
    const int i  = blockIdx.x * 256 + threadIdx.x;
    const int NC = NGATE * KENC;
    if (i < NC) {
        const int n = i / KENC, k = i - n * KENC;
        const float v = (k < H_DIM) ? Whh_e[n * H_DIM + k] : Wih_e[n * I_DIM + (k - H_DIM)];
        wcat[i] = (bf16_t)v;
    } else {
        const int jv = i - NC;
        if (jv < NGATE * H_DIM) wdec[jv] = (bf16_t)Whh_d[jv];
    }
}

// ---------------------------------------------------------------------------
// Main persistent kernel: 32 workgroups x 256 threads, each owns 32 batch rows
// through encoder scan, latent block, and decoder scan.
// ---------------------------------------------------------------------------
extern "C" __global__ void __launch_bounds__(256)
vae_main(const float* __restrict__ src, const float* __restrict__ eps,
         const float* __restrict__ b_e,
         const float* __restrict__ W_mu, const float* __restrict__ b_mu,
         const float* __restrict__ W_lv, const float* __restrict__ b_lv,
         const float* __restrict__ W_zh, const float* __restrict__ b_zh,
         const float* __restrict__ W_zc, const float* __restrict__ b_zc,
         const float* __restrict__ W_ih_d, const float* __restrict__ b_d,
         const float* __restrict__ W_out, const float* __restrict__ b_out,
         const bf16_t* __restrict__ Wcat, const bf16_t* __restrict__ Wdec,
         float* __restrict__ out)
{
    extern __shared__ __align__(16) char smem[];
    bf16_t* aS    = (bf16_t*)(smem + OFF_A);
    float*  cS    = (float*)(smem + OFF_C);
    float*  biasS = (float*)(smem + OFF_BIAS);
    float*  wihS  = (float*)(smem + OFF_WIH);
    float*  woutS = (float*)(smem + OFF_WOUT);
    float*  zS    = (float*)(smem + OFF_Z);
    float*  xS    = (float*)(smem + OFF_X);
    float*  redS  = (float*)(smem + OFF_RED);

    const int tid  = threadIdx.x;
    const int wave = tid >> 5;
    const int lane = tid & 31;
    const int b0   = blockIdx.x * MB;

    // ---- init: h=0 (both buffers), c=0, encoder bias to LDS, stage x_0 ----
    for (int i = tid; i < 2 * ABUF; i += 256) aS[i] = (bf16_t)0.0f;
    for (int i = tid; i < MB * CST; i += 256) cS[i] = 0.0f;
    for (int i = tid; i < NGATE;    i += 256) biasS[i] = b_e[i];
    __syncthreads();
    load_x(src, aS, b0, 0);
    __syncthreads();

    // ---- encoder scan: one barrier per step (double-buffered h||x) ----
    for (int t = 0; t < T_SRC; ++t) {
        const bf16_t* aCur = aS + (t & 1) * ABUF;
        bf16_t*       aNxt = aS + ((t + 1) & 1) * ABUF;
        if (t + 1 < T_SRC) load_x(src, aNxt, b0, t + 1);   // overlap with WMMAs
        lstm_step<KENC / 32, false>(Wcat, KENC, aCur, aNxt, cS, biasS,
                                    nullptr, nullptr, wave, lane);
        __syncthreads();
    }

    // ---- latent block: mu / logvar / z (h_T lives in buffer 0) ----
    const bf16_t* hN = aS;
    {
        const int m  = tid >> 3;
        const int li = tid & 7;
        float accMu[8], accLv[8];
        #pragma unroll
        for (int s = 0; s < 8; ++s) { accMu[s] = b_mu[li * 8 + s]; accLv[s] = b_lv[li * 8 + s]; }
        for (int jj = 0; jj < H_DIM; ++jj) {
            const float hv = (float)hN[m * LDA + jj];
            #pragma unroll
            for (int s = 0; s < 8; ++s) {
                const int l = li * 8 + s;
                accMu[s] += hv * W_mu[l * H_DIM + jj];
                accLv[s] += hv * W_lv[l * H_DIM + jj];
            }
        }
        #pragma unroll
        for (int s = 0; s < 8; ++s) {
            const int l  = li * 8 + s;
            const float mu = accMu[s], lv = accLv[s];
            out[(size_t)(B_TOT * TGT_LEN)                 + (size_t)(b0 + m) * L_DIM + l] = mu;
            out[(size_t)(B_TOT * TGT_LEN + B_TOT * L_DIM) + (size_t)(b0 + m) * L_DIM + l] = lv;
            zS[m * L_DIM + l] = mu + eps[(size_t)(b0 + m) * L_DIM + l] * __expf(0.5f * lv);
        }
    }
    __syncthreads();

    // ---- z -> (h0, c0); load decoder constants ----
    {
        const int m   = tid >> 3;
        const int blk = tid & 7;
        for (int s = 0; s < 64; ++s) {
            const int jj = blk * 64 + s;
            float ah = b_zh[jj], ac = b_zc[jj];
            for (int l = 0; l < L_DIM; ++l) {
                const float zv = zS[m * L_DIM + l];
                ah += zv * W_zh[jj * L_DIM + l];
                ac += zv * W_zc[jj * L_DIM + l];
            }
            aS[m * LDA + jj] = (bf16_t)tanh_f(ah);   // buffer 0 = decoder aCur at t=0
            cS[m * CST + jj] = tanh_f(ac);
        }
    }
    for (int i = tid; i < NGATE; i += 256) { biasS[i] = b_d[i]; wihS[i] = W_ih_d[i]; }
    for (int i = tid; i < H_DIM; i += 256) woutS[i] = W_out[i];
    const float bout0 = b_out[0];
    if (tid < MB) { xS[tid] = 0.0f; redS[tid] = bout0; }
    __syncthreads();

    // ---- autoregressive decoder scan ----
    for (int t = 0; t < TGT_LEN; ++t) {
        const bf16_t* aCur = aS + (t & 1) * ABUF;
        bf16_t*       aNxt = aS + ((t + 1) & 1) * ABUF;
        lstm_step<H_DIM / 32, true>(Wdec, H_DIM, aCur, aNxt, cS, biasS,
                                    wihS, xS, wave, lane);
        __syncthreads();

        // fc_out: out[m] = h_new[m,:] . W_out + b_out  (8 threads per batch row)
        {
            const int m = tid >> 3, seg = tid & 7;
            const bf16_t* hrow = aNxt + m * LDA + seg * 64;
            const float*  wrow = woutS + seg * 64;
            float s = 0.0f;
            for (int k2 = 0; k2 < 64; ++k2) s += (float)hrow[k2] * wrow[k2];
            atomicAdd(&redS[m], s);                  // ds_add_f32
        }
        __syncthreads();

        if (tid < MB) {
            const float o = redS[tid];
            out[(size_t)(b0 + tid) * TGT_LEN + t] = o;   // recon[b][t]
            xS[tid]   = o;                                // feedback
            redS[tid] = bout0;                            // re-arm reduction
        }
        __syncthreads();
    }
}

// ---------------------------------------------------------------------------

extern "C" void kernel_launch(void* const* d_in, const int* in_sizes, int n_in,
                              void* d_out, int out_size, void* d_ws, size_t ws_size,
                              hipStream_t stream)
{
    (void)in_sizes; (void)n_in; (void)out_size; (void)ws_size;
    const float* src   = (const float*)d_in[0];
    const float* eps   = (const float*)d_in[1];
    // d_in[2] = tgt_len (== 1024, compiled in)
    const float* Wih_e = (const float*)d_in[3];
    const float* Whh_e = (const float*)d_in[4];
    const float* b_e   = (const float*)d_in[5];
    const float* W_mu  = (const float*)d_in[6];
    const float* b_mu  = (const float*)d_in[7];
    const float* W_lv  = (const float*)d_in[8];
    const float* b_lv  = (const float*)d_in[9];
    const float* W_zh  = (const float*)d_in[10];
    const float* b_zh  = (const float*)d_in[11];
    const float* W_zc  = (const float*)d_in[12];
    const float* b_zc  = (const float*)d_in[13];
    const float* Wih_d = (const float*)d_in[14];
    const float* Whh_d = (const float*)d_in[15];
    const float* b_d   = (const float*)d_in[16];
    const float* W_out = (const float*)d_in[17];
    const float* b_out = (const float*)d_in[18];

    bf16_t* wcat = (bf16_t*)d_ws;                                            // 2048x576 bf16
    bf16_t* wdec = (bf16_t*)((char*)d_ws + (size_t)NGATE * KENC * sizeof(bf16_t)); // 2048x512 bf16

    const int prepTotal = NGATE * KENC + NGATE * H_DIM;
    vae_prep<<<(prepTotal + 255) / 256, 256, 0, stream>>>(Wih_e, Whh_e, Whh_d, wcat, wdec);

    (void)hipFuncSetAttribute((const void*)vae_main,
                              hipFuncAttributeMaxDynamicSharedMemorySize, SMEM_BYTES);
    vae_main<<<B_TOT / MB, 256, SMEM_BYTES, stream>>>(
        src, eps, b_e, W_mu, b_mu, W_lv, b_lv, W_zh, b_zh, W_zc, b_zc,
        Wih_d, b_d, W_out, b_out, wcat, wdec, (float*)d_out);
}
